// CriticNetwork_84911503442379
// MI455X (gfx1250) — compile-verified, matrix operand order checked by reading, not run on hardware
//
#include <hip/hip_runtime.h>
#include <hip/hip_bf16.h>

typedef __attribute__((ext_vector_type(16))) __bf16     v16bf;
typedef __attribute__((ext_vector_type(4)))  __bf16     v4bf;
typedef __attribute__((ext_vector_type(8)))  float      v8f;
typedef __attribute__((ext_vector_type(4)))  unsigned   u32x4;
typedef __attribute__((ext_vector_type(4)))  float      f32x4;

#define BDIM 2048
#define BM 128
#define BN 128
#define BK 32
#define LDSK (BK + 8)   // padded ushort stride to spread LDS banks

// Pack 8 floats -> 8 bf16 (one 16B LDS word). __builtin_convertvector lets the
// backend pick v_cvt_pk_bf16_f32 / v_cvt_bf16_f32 when available; its generic
// expansion is the same RNE bit sequence otherwise.
__device__ __forceinline__ u32x4 pack8_bf16(const float r[8]) {
    f32x4 lo = {r[0], r[1], r[2], r[3]};
    f32x4 hi = {r[4], r[5], r[6], r[7]};
    union { v4bf h[2]; u32x4 u; } cv;
    cv.h[0] = __builtin_convertvector(lo, v4bf);
    cv.h[1] = __builtin_convertvector(hi, v4bf);
    return cv.u;
}

// Branch-free full-K tile load: rows clamped (garbage rows feed only masked
// outputs), always two global_load_b128 per 64-row half.
__device__ __forceinline__ void load_tile_fast(
    const float* __restrict__ base, int row0, int K, int k0,
    int ldRow, int ldCol, int maxRow, float r0[8], float r1[8])
{
    int g0 = row0 + ldRow;       if (g0 > maxRow) g0 = maxRow;
    int g1 = row0 + 64 + ldRow;  if (g1 > maxRow) g1 = maxRow;
    const float* s0 = base + (long long)g0 * K + k0 + ldCol;
    const float* s1 = base + (long long)g1 * K + k0 + ldCol;
    f32x4 v00 = *(const f32x4*)(s0);
    f32x4 v01 = *(const f32x4*)(s0 + 4);
    f32x4 v10 = *(const f32x4*)(s1);
    f32x4 v11 = *(const f32x4*)(s1 + 4);
#pragma unroll
    for (int j = 0; j < 4; ++j) {
        r0[j] = v00[j]; r0[j + 4] = v01[j];
        r1[j] = v10[j]; r1[j + 4] = v11[j];
    }
}

// K-tail tile load: zero-fill OOB K (those would pollute valid accumulators).
__device__ __forceinline__ void load_tile_tail(
    const float* __restrict__ base, int row0, int K, int k0,
    int ldRow, int ldCol, int maxRow, float r0[8], float r1[8])
{
    int g0 = row0 + ldRow;       if (g0 > maxRow) g0 = maxRow;
    int g1 = row0 + 64 + ldRow;  if (g1 > maxRow) g1 = maxRow;
#pragma unroll
    for (int j = 0; j < 8; ++j) {
        int gk = k0 + ldCol + j;
        bool ok = gk < K;
        r0[j] = ok ? base[(long long)g0 * K + gk] : 0.f;
        r1[j] = ok ? base[(long long)g1 * K + gk] : 0.f;
    }
}

__device__ __forceinline__ void store_tile_lds(
    unsigned short* __restrict__ dst, int ldRow, int ldCol,
    const float r0[8], const float r1[8])
{
    *(u32x4*)&dst[ldRow * LDSK + ldCol]        = pack8_bf16(r0);
    *(u32x4*)&dst[(64 + ldRow) * LDSK + ldCol] = pack8_bf16(r1);
}

// ---------------------------------------------------------------------------
// C[M,N] = A[M,K] @ W[N,K]^T + bias[N]; batched over blockIdx.z with strides.
// bf16 WMMA, fp32 accumulate, double-buffered LDS.
// ---------------------------------------------------------------------------
__global__ __launch_bounds__(256) void gemm_bf16_wmma(
    const float* __restrict__ Ag, const float* __restrict__ Wg,
    const float* __restrict__ biasg, float* __restrict__ Cg,
    int M, int N, int K,
    long long sA, long long sW, long long sBias, long long sC)
{
    const int z = blockIdx.z;
    const float* A    = Ag + (long long)z * sA;
    const float* W    = Wg + (long long)z * sW;
    const float* bias = biasg + (long long)z * sBias;
    float*       C    = Cg + (long long)z * sC;

    __shared__ unsigned short lA[2][BM * LDSK];
    __shared__ unsigned short lB[2][BN * LDSK];

    const int tid   = threadIdx.x;
    const int lane  = tid & 31;
    const int wave  = tid >> 5;
    const int waveM = wave >> 2;   // 0..1
    const int waveN = wave & 3;    // 0..3
    const int m0 = blockIdx.y * BM;
    const int n0 = blockIdx.x * BN;

    v8f acc[4][2];
#pragma unroll
    for (int i = 0; i < 4; ++i)
#pragma unroll
        for (int j = 0; j < 2; ++j)
            acc[i][j] = (v8f){0.f,0.f,0.f,0.f,0.f,0.f,0.f,0.f};

    const int ldRow = tid >> 2;          // 0..63
    const int ldCol = (tid & 3) * 8;     // 0,8,16,24

    const int nFull = K / BK;
    const int nIter = (K + BK - 1) / BK;

    float ra0[8], ra1[8], rb0[8], rb1[8];

    // ---- prologue: stage tile 0 into buffer 0 ----
    if (nFull > 0) {
        load_tile_fast(A, m0, K, 0, ldRow, ldCol, M - 1, ra0, ra1);
        load_tile_fast(W, n0, K, 0, ldRow, ldCol, N - 1, rb0, rb1);
    } else {
        load_tile_tail(A, m0, K, 0, ldRow, ldCol, M - 1, ra0, ra1);
        load_tile_tail(W, n0, K, 0, ldRow, ldCol, N - 1, rb0, rb1);
    }
    store_tile_lds(lA[0], ldRow, ldCol, ra0, ra1);
    store_tile_lds(lB[0], ldRow, ldCol, rb0, rb1);
    __syncthreads();

    const int kb   = (lane >> 4) * 8;   // ushort offset 0 or 8
    const int lrow = lane & 15;

    for (int it = 0; it < nIter; ++it) {
        const int cb = it & 1;
        const bool hasNext = (it + 1) < nIter;

        // ---- issue next tile's global loads first (overlap with WMMA) ----
        if (hasNext) {
            int k0n = (it + 1) * BK;
            if (it + 1 < nFull) {
                load_tile_fast(A, m0, K, k0n, ldRow, ldCol, M - 1, ra0, ra1);
                load_tile_fast(W, n0, K, k0n, ldRow, ldCol, N - 1, rb0, rb1);
            } else {
                load_tile_tail(A, m0, K, k0n, ldRow, ldCol, M - 1, ra0, ra1);
                load_tile_tail(W, n0, K, k0n, ldRow, ldCol, N - 1, rb0, rb1);
            }
            if (it + 2 < nFull) {
                int g = m0 + ldRow; if (g > M - 1) g = M - 1;
                __builtin_prefetch(A + (long long)g * K + (it + 2) * BK + ldCol, 0, 3);
                g = n0 + ldRow; if (g > N - 1) g = N - 1;
                __builtin_prefetch(W + (long long)g * K + (it + 2) * BK + ldCol, 0, 3);
            }
        }

        // ---- fragments from current buffer + 8 WMMAs ----
        union FragU { u32x4 q[2]; v16bf v; };
        v16bf af[4], bf_[2];
#pragma unroll
        for (int tn = 0; tn < 2; ++tn) {
            int row = waveN * 32 + tn * 16 + lrow;
            FragU fu;
            fu.q[0] = *(const u32x4*)&lB[cb][row * LDSK + kb];
            fu.q[1] = *(const u32x4*)&lB[cb][row * LDSK + kb + 16];
            bf_[tn] = fu.v;
        }
#pragma unroll
        for (int tm = 0; tm < 4; ++tm) {
            int row = waveM * 64 + tm * 16 + lrow;
            FragU fu;
            fu.q[0] = *(const u32x4*)&lA[cb][row * LDSK + kb];
            fu.q[1] = *(const u32x4*)&lA[cb][row * LDSK + kb + 16];
            af[tm] = fu.v;
        }
#pragma unroll
        for (int tm = 0; tm < 4; ++tm)
#pragma unroll
            for (int tn = 0; tn < 2; ++tn)
                acc[tm][tn] = __builtin_amdgcn_wmma_f32_16x16x32_bf16(
                    false, af[tm], false, bf_[tn], (short)0, acc[tm][tn],
                    false, false);

        // ---- store next tile into other buffer ----
        if (hasNext) {
            store_tile_lds(lA[cb ^ 1], ldRow, ldCol, ra0, ra1);
            store_tile_lds(lB[cb ^ 1], ldRow, ldCol, rb0, rb1);
        }
        __syncthreads();
    }

    // ---- epilogue: C layout: VGPR r -> M = r + 8*(lane>=16); N = lane&15 ----
#pragma unroll
    for (int tn = 0; tn < 2; ++tn) {
        int col = n0 + waveN * 32 + tn * 16 + (lane & 15);
        float bv = (col < N) ? bias[col] : 0.f;
#pragma unroll
        for (int tm = 0; tm < 4; ++tm) {
            int rowBase = m0 + waveM * 64 + tm * 16 + (lane >> 4) * 8;
#pragma unroll
            for (int r = 0; r < 8; ++r) {
                int row = rowBase + r;
                if (row < M && col < N)
                    C[(long long)row * N + col] = acc[tm][tn][r] + bv;
            }
        }
    }
}

// ---------------------------------------------------------------------------
// Attention: logits -> softmax -> mask -> weighted neighbor sum.
// Z: [8,B,cur], Wa: [8,7,7,cur], ab: [8,7], Amat: [B,8,8], neigh: [8,B,cur]
// ---------------------------------------------------------------------------
__global__ __launch_bounds__(128) void attn_kernel(
    const float* __restrict__ Z, const float* __restrict__ Wa,
    const float* __restrict__ ab, const float* __restrict__ Amat,
    float* __restrict__ neigh, int B, int cur)
{
    const int k = blockIdx.y;
    const int b = blockIdx.x * blockDim.x + threadIdx.x;
    if (b >= B) return;

    int nbr[7];
#pragma unroll
    for (int m = 0; m < 7; ++m) nbr[m] = (m < k) ? m : m + 1;

    float mask[7];
#pragma unroll
    for (int m = 0; m < 7; ++m) {
        int ai, aj;
        if (k == 0) { ai = (m < 3) ? 0 : (m + 1); aj = (m < 3) ? (m + 1) : 0; }
        else        { ai = nbr[m]; aj = k; }
        mask[m] = Amat[((long long)b * 8 + ai) * 8 + aj];
    }

    float logits[7];
#pragma unroll
    for (int o = 0; o < 7; ++o) logits[o] = ab[k * 7 + o];

    for (int m = 0; m < 7; ++m) {
        const float* zrow = Z  + ((long long)(nbr[m] * B + b)) * cur;
        const float* wbas = Wa + ((long long)k * 49 + m) * cur;
        for (int d = 0; d < cur; ++d) {
            float zv = zrow[d];
#pragma unroll
            for (int o = 0; o < 7; ++o)
                logits[o] += zv * wbas[(long long)o * 7 * cur + d];
        }
    }

    float mx = logits[0];
#pragma unroll
    for (int o = 1; o < 7; ++o) mx = fmaxf(mx, logits[o]);
    float sum = 0.f, w[7];
#pragma unroll
    for (int o = 0; o < 7; ++o) { w[o] = __expf(logits[o] - mx); sum += w[o]; }
    float inv = 1.f / sum;
#pragma unroll
    for (int o = 0; o < 7; ++o) w[o] = w[o] * inv * mask[o];

    const float* zp[7];
#pragma unroll
    for (int m = 0; m < 7; ++m) zp[m] = Z + ((long long)(nbr[m] * B + b)) * cur;
    float* out = neigh + ((long long)(k * B + b)) * cur;
    for (int d = 0; d < cur; ++d) {
        float a = 0.f;
#pragma unroll
        for (int m = 0; m < 7; ++m) a += w[m] * zp[m][d];
        out[d] = a;
    }
}

// ---------------------------------------------------------------------------
// GRU elementwise: gi/gh [rows,3*cur], Z,h [rows,cur]; rows = 8*B
// ---------------------------------------------------------------------------
__global__ __launch_bounds__(256) void gru_kernel(
    const float* __restrict__ gi, const float* __restrict__ gh,
    const float* __restrict__ Z, float* __restrict__ h,
    long long total, int cur)
{
    long long t = (long long)blockIdx.x * blockDim.x + threadIdx.x;
    if (t >= total) return;
    long long row = t / cur;
    int d = (int)(t - row * cur);
    const float* pi = gi + row * 3LL * cur;
    const float* ph = gh + row * 3LL * cur;
    float r  = 1.f / (1.f + __expf(-(pi[d] + ph[d])));
    float zg = 1.f / (1.f + __expf(-(pi[cur + d] + ph[cur + d])));
    float n  = tanhf(pi[2 * cur + d] + r * ph[2 * cur + d]);
    h[t] = (1.f - zg) * n + zg * Z[t];
}

// ---------------------------------------------------------------------------
// LayerNorm (+ ReLU). One block per row. Gamma/beta group = row/rowsPerGroup.
// ---------------------------------------------------------------------------
__global__ __launch_bounds__(256) void ln_relu_kernel(
    const float* __restrict__ X, const float* __restrict__ G,
    const float* __restrict__ Bt, float* __restrict__ Y,
    int n, int rowsPerGroup, int doRelu)
{
    __shared__ float s[256];
    const int row = blockIdx.x;
    const float* x = X + (long long)row * n;
    float*       y = Y + (long long)row * n;
    const int grp = row / rowsPerGroup;
    const float* g  = G  + (long long)grp * n;
    const float* bb = Bt + (long long)grp * n;
    const int tid = threadIdx.x;

    float sum = 0.f;
    for (int i = tid; i < n; i += 256) sum += x[i];
    s[tid] = sum; __syncthreads();
    for (int o = 128; o > 0; o >>= 1) { if (tid < o) s[tid] += s[tid + o]; __syncthreads(); }
    float mu = s[0] / n;
    __syncthreads();

    float vs = 0.f;
    for (int i = tid; i < n; i += 256) { float d = x[i] - mu; vs += d * d; }
    s[tid] = vs; __syncthreads();
    for (int o = 128; o > 0; o >>= 1) { if (tid < o) s[tid] += s[tid + o]; __syncthreads(); }
    float inv = rsqrtf(s[0] / n + 1e-5f);
    __syncthreads();

    for (int i = tid; i < n; i += 256) {
        float v = (x[i] - mu) * inv * g[i] + bb[i];
        if (doRelu) v = fmaxf(v, 0.f);
        y[i] = v;
    }
}

// ---------------------------------------------------------------------------
// Head concat: flat[b, 0:8192]=Z[k,b,d]; [8192:8224]=state; [8224:8232]=action
// ---------------------------------------------------------------------------
__global__ __launch_bounds__(256) void concat_kernel(
    const float* __restrict__ Z, const float* __restrict__ st,
    const float* __restrict__ ac, float* __restrict__ flat, int B)
{
    const int KF = 8232;
    long long idx = (long long)blockIdx.x * blockDim.x + threadIdx.x;
    if (idx >= (long long)B * KF) return;
    int b = (int)(idx / KF);
    int c = (int)(idx - (long long)b * KF);
    float v;
    if (c < 8192) {
        int k = c >> 10, d = c & 1023;
        v = Z[((long long)k * B + b) * 1024 + d];
    } else if (c < 8224) {
        v = st[b * 32 + (c - 8192)];
    } else {
        v = ac[b * 8 + (c - 8224)];
    }
    flat[idx] = v;
}

// Final q4: out[b] = dot(q[b,0:64], w) + b0
__global__ __launch_bounds__(256) void q4_kernel(
    const float* __restrict__ q, const float* __restrict__ w,
    const float* __restrict__ b0, float* __restrict__ out, int B)
{
    int b = blockIdx.x * blockDim.x + threadIdx.x;
    if (b >= B) return;
    float a = b0[0];
#pragma unroll
    for (int i = 0; i < 64; ++i) a += q[(long long)b * 64 + i] * w[i];
    out[b] = a;
}

static inline int cdiv(int a, int b) { return (a + b - 1) / b; }

extern "C" void kernel_launch(void* const* d_in, const int* in_sizes, int n_in,
                              void* d_out, int out_size, void* d_ws, size_t ws_size,
                              hipStream_t stream) {
    const int B = BDIM;
    const float* state  = (const float*)d_in[0];
    const float* action = (const float*)d_in[1];
    const float* Amat   = (const float*)d_in[10];
    // params flattened in jax tree_leaves order (dict keys sorted)
    const float* b1_b = (const float*)d_in[11];
    const float* b1_g = (const float*)d_in[12];
    const float* b2_b = (const float*)d_in[13];
    const float* b2_g = (const float*)d_in[14];
    const float* b3_b = (const float*)d_in[15];
    const float* b3_g = (const float*)d_in[16];
    const float* q1_b = (const float*)d_in[17];
    const float* q1_w = (const float*)d_in[18];
    const float* q2_b = (const float*)d_in[19];
    const float* q2_w = (const float*)d_in[20];
    const float* q3_b = (const float*)d_in[21];
    const float* q3_w = (const float*)d_in[22];
    const float* q4_b = (const float*)d_in[23];
    const float* q4_w = (const float*)d_in[24];

    float* ws = (float*)d_ws;
    float* bufZ  = ws;
    float* bufN  = bufZ  + (size_t)8 * B * 1024;
    float* bufG1 = bufN  + (size_t)8 * B * 1024;
    float* bufG2 = bufG1 + (size_t)8 * B * 3072;
    float* bufF  = bufG2 + (size_t)8 * B * 3072;

    // pack Z1..Z8 -> bufZ [8,B,60]
    for (int i = 0; i < 8; ++i)
        hipMemcpyAsync(bufZ + (size_t)i * B * 60, d_in[2 + i],
                       (size_t)B * 60 * sizeof(float),
                       hipMemcpyDeviceToDevice, stream);

    const int curs[3] = {60, 1024, 1024};
    for (int s = 0; s < 3; ++s) {
        const int base = 25 + s * 10;
        const float* att_b   = (const float*)d_in[base + 0];
        const float* att_w   = (const float*)d_in[base + 1];
        const float* gru_bhh = (const float*)d_in[base + 2];
        const float* gru_bih = (const float*)d_in[base + 3];
        const float* gru_whh = (const float*)d_in[base + 4];
        const float* gru_wih = (const float*)d_in[base + 5];
        const float* lin_b   = (const float*)d_in[base + 6];
        const float* lin_w   = (const float*)d_in[base + 7];
        const float* ln_b    = (const float*)d_in[base + 8];
        const float* ln_g    = (const float*)d_in[base + 9];
        const int cur = curs[s];
        const int tc  = 3 * cur;

        // attention -> bufN (neigh)
        attn_kernel<<<dim3(cdiv(B, 128), 8), 128, 0, stream>>>(
            bufZ, att_w, att_b, Amat, bufN, B, cur);

        // gi = neigh @ wih^T + bih -> bufG1 ; gh = Z @ whh^T + bhh -> bufG2
        gemm_bf16_wmma<<<dim3(cdiv(tc, BN), cdiv(B, BM), 8), 256, 0, stream>>>(
            bufN, gru_wih, gru_bih, bufG1, B, tc, cur,
            (long long)B * cur, (long long)tc * cur, tc, (long long)B * tc);
        gemm_bf16_wmma<<<dim3(cdiv(tc, BN), cdiv(B, BM), 8), 256, 0, stream>>>(
            bufZ, gru_whh, gru_bhh, bufG2, B, tc, cur,
            (long long)B * cur, (long long)tc * cur, tc, (long long)B * tc);

        // GRU elementwise -> h in bufN (overwrites neigh)
        long long tot = (long long)8 * B * cur;
        gru_kernel<<<(int)((tot + 255) / 256), 256, 0, stream>>>(
            bufG1, bufG2, bufZ, bufN, tot, cur);

        // y = h @ lin_w^T + lin_b -> bufG1 [8,B,1024]
        gemm_bf16_wmma<<<dim3(cdiv(1024, BN), cdiv(B, BM), 8), 256, 0, stream>>>(
            bufN, lin_w, lin_b, bufG1, B, 1024, cur,
            (long long)B * cur, (long long)1024 * cur, 1024, (long long)B * 1024);

        // LN + ReLU -> bufZ (next Z, cur=1024)
        ln_relu_kernel<<<8 * B, 256, 0, stream>>>(
            bufG1, ln_g, ln_b, bufZ, 1024, B, 1);
    }

    // head
    const int KF = 8232;
    long long ctot = (long long)B * KF;
    concat_kernel<<<(int)((ctot + 255) / 256), 256, 0, stream>>>(
        bufZ, state, action, bufF, B);

    float* q1o = bufN;
    float* q2o = bufN + (size_t)B * 256;
    float* q3o = q2o  + (size_t)B * 128;

    gemm_bf16_wmma<<<dim3(cdiv(256, BN), cdiv(B, BM), 1), 256, 0, stream>>>(
        bufF, q1_w, q1_b, q1o, B, 256, KF, 0, 0, 0, 0);
    ln_relu_kernel<<<B, 256, 0, stream>>>(q1o, b1_g, b1_b, q1o, 256, B, 1);

    gemm_bf16_wmma<<<dim3(cdiv(128, BN), cdiv(B, BM), 1), 256, 0, stream>>>(
        q1o, q2_w, q2_b, q2o, B, 128, 256, 0, 0, 0, 0);
    ln_relu_kernel<<<B, 256, 0, stream>>>(q2o, b2_g, b2_b, q2o, 128, B, 1);

    gemm_bf16_wmma<<<dim3(cdiv(64, BN), cdiv(B, BM), 1), 256, 0, stream>>>(
        q2o, q3_w, q3_b, q3o, B, 64, 128, 0, 0, 0, 0);
    ln_relu_kernel<<<B, 256, 0, stream>>>(q3o, b3_g, b3_b, q3o, 64, B, 1);

    q4_kernel<<<cdiv(B, 256), 256, 0, stream>>>(q3o, q4_w, q4_b, (float*)d_out, B);
}
